// P2I_CrossAttention_21139829030939
// MI455X (gfx1250) — compile-verified, matrix operand order checked by reading, not run on hardware
//
#include <hip/hip_runtime.h>
#include <hip/hip_bf16.h>
#include <stdint.h>

// P2I cross-attention for MI455X (gfx1250), bf16 WMMA pipeline.
// B=8, C=128, WH=48*48=2304, N=8192.

#define B_    8
#define C_    128
#define WH_   2304
#define N_    8192
#define PITCH 136   // bf16 elements per LDS row (128 + 8 pad; 272B, 16B-aligned)
#define VPITCH 72   // bf16 elements per LDS row for 64-wide V staging (144B)
#define OPITCH 132  // fp32 elements per LDS row for 128-wide out staging (528B)

typedef __attribute__((ext_vector_type(16))) __bf16 bf16x16;
typedef __attribute__((ext_vector_type(8)))  float  v8f;
typedef __attribute__((ext_vector_type(4)))  int    v4i_;

#if defined(__has_builtin)
#if __has_builtin(__builtin_amdgcn_global_load_async_to_lds_b128)
#define HAS_ASYNC_LDS 1
#pragma message("gfx1250: using GLOBAL_LOAD_ASYNC_TO_LDS_B128 path")
#else
#define HAS_ASYNC_LDS 0
#pragma message("gfx1250: async-to-LDS builtin NOT available, using sync copies")
#endif
#else
#define HAS_ASYNC_LDS 0
#endif

__device__ __forceinline__ float sigmoidf_(float x) { return 1.0f / (1.0f + __expf(-x)); }

// One b128 global->LDS copy; async (ASYNCcnt-tracked) when available.
__device__ __forceinline__ void copy_b128(const void* g, void* l) {
#if HAS_ASYNC_LDS
  __builtin_amdgcn_global_load_async_to_lds_b128(
      (__attribute__((address_space(1))) v4i_*)g,
      (__attribute__((address_space(3))) v4i_*)l, 0, 0);
#else
  *(uint4*)l = *(const uint4*)g;
#endif
}
__device__ __forceinline__ void async_wait_all() {
#if HAS_ASYNC_LDS
  asm volatile("s_wait_asynccnt 0x0" ::: "memory");
#endif
}
// Wait until at most the 16 newest async ops are outstanding (1 chunk = 16/wave).
__device__ __forceinline__ void async_wait_16() {
#if HAS_ASYNC_LDS
  asm volatile("s_wait_asynccnt 0x10" ::: "memory");
#endif
}

// Load one 16-bit WMMA A/B fragment from a row-major bf16 tile.
// Per CDNA5 ISA 16-bit layout: lanes 0-15 take K = k0+[0..7] and k0+16+[0..7];
// lanes 16-31 take K = k0+8+[0..7] and k0+24+[0..7] -> two b128 LDS loads.
__device__ __forceinline__ bf16x16 ld_frag(const __bf16* rowbase, int k0, int lane) {
  const __bf16* p = rowbase + k0 + ((lane & 16) ? 8 : 0);
  union { bf16x16 v; uint4 q[2]; } u;
  u.q[0] = *reinterpret_cast<const uint4*>(p);
  u.q[1] = *reinterpret_cast<const uint4*>(p + 16);
  return u.v;
}

__device__ __forceinline__ v8f wmma_bf16(bf16x16 a, bf16x16 b, v8f c) {
  return __builtin_amdgcn_wmma_f32_16x16x32_bf16(false, a, false, b, (short)0, c, false, false);
}

// 16x128x128 GEMM strip: res[ct] (+)= A(16x128) * B_tile(ct).
// A fragments preloaded by caller; B fragments pipelined one step deep.
template <bool ACCUM>
__device__ __forceinline__ void gemm_strip(const bf16x16 (&af)[4], const __bf16* B,
                                           int nl, int lane, v8f (&res)[8])
{
  const v8f vzero = {};
  const __bf16* row0 = B + nl * PITCH;
  bf16x16 bbuf[2];
  bbuf[0] = ld_frag(row0, 0, lane);
  v8f acc = ACCUM ? res[0] : vzero;
#pragma unroll
  for (int i = 0; i < 32; ++i) {
    const int ct = i >> 2, kc = i & 3;
    if (i + 1 < 32) {
      const int nct = (i + 1) >> 2, nkc = (i + 1) & 3;
      bbuf[(i + 1) & 1] = ld_frag(row0 + nct * 16 * PITCH, nkc * 32, lane);
    }
    acc = wmma_bf16(af[kc], bbuf[i & 1], acc);
    if (kc == 3) {
      res[ct] = acc;
      if (i + 1 < 32) acc = ACCUM ? res[ct + 1] : vzero;
    }
  }
}

// ---------------------------------------------------------------------------
// Kernel 1: pq = feat0^T @ Wq + bq ; geo = geo_embed @ Wg + bg ;
//           gate = sigmoid([pq,geo] @ Wgate + bgate) ; pq += gate*geo.
// ---------------------------------------------------------------------------
__global__ void proj_qgeo_kernel(const float* __restrict__ feat0,
                                 const float* __restrict__ geo_embed,
                                 const float* __restrict__ Wq, const float* __restrict__ bq,
                                 const float* __restrict__ Wg, const float* __restrict__ bg,
                                 const float* __restrict__ Wgate, const float* __restrict__ bgate,
                                 __bf16* __restrict__ pq_bf, float* __restrict__ geoT)
{
  extern __shared__ __align__(16) char smem[];
  __bf16* sWq  = (__bf16*)smem;               // [128][PITCH], transposed: sWq[n][k]
  __bf16* sWg  = sWq + C_ * PITCH;
  __bf16* sA   = sWg + C_ * PITCH;            // q_in tile  [64][PITCH]
  __bf16* sG   = sA  + 64 * PITCH;            // geo tile   [64][PITCH]
  float*  sPQ  = (float*)(sG + 64 * PITCH);   // [64][128] fp32
  float*  sGE  = sPQ + 64 * C_;               // [64][128] fp32
  float*  sGate = sGE + 64 * C_;              // [64]

  const int tid = threadIdx.x;
  const int b   = blockIdx.x / (WH_ / 64);
  const int p0  = (blockIdx.x % (WH_ / 64)) * 64;

  for (int idx = tid; idx < C_ * C_; idx += 256) {
    int k = idx >> 7, n = idx & 127;
    sWq[n * PITCH + k] = (__bf16)Wq[idx];
    sWg[n * PITCH + k] = (__bf16)Wg[idx];
  }
  for (int idx = tid; idx < 64 * C_; idx += 256) {
    int c = idx >> 6, m = idx & 63;
    sA[m * PITCH + c] = (__bf16)feat0[((size_t)b * C_ + c) * WH_ + (p0 + m)];
  }
  for (int idx = tid; idx < 64 * C_; idx += 256) {
    int m = idx >> 7, c = idx & 127;
    sG[m * PITCH + c] = (__bf16)geo_embed[((size_t)b * WH_ + p0 + m) * C_ + c];
  }

  const int w = tid >> 5, lane = tid & 31, nl = lane & 15;
  const __bf16* Amat = (w < 4) ? sA  : sG;
  const __bf16* Wmat = (w < 4) ? sWq : sWg;
  float*        Omat = (w < 4) ? sPQ : sGE;
  const float*  bias = (w < 4) ? bq  : bg;
  const int mrow = (w & 3) * 16;
  const int mb = mrow + ((lane & 16) ? 8 : 0);

  float bb[8];
#pragma unroll
  for (int ct = 0; ct < 8; ++ct) bb[ct] = bias[ct * 16 + nl];

  __syncthreads();

  const __bf16* arow = Amat + (mrow + nl) * PITCH;
  bf16x16 afrag[4];
#pragma unroll
  for (int kc = 0; kc < 4; ++kc) afrag[kc] = ld_frag(arow, kc * 32, lane);

  v8f res[8];
  gemm_strip<false>(afrag, Wmat, nl, lane, res);
#pragma unroll
  for (int ct = 0; ct < 8; ++ct)
#pragma unroll
    for (int j = 0; j < 8; ++j)
      Omat[(mb + j) * C_ + ct * 16 + nl] = res[ct][j] + bb[ct];
  __syncthreads();

  if (tid < 64) {
    float s = bgate[0];
    for (int c = 0; c < C_; ++c) s += sPQ[tid * C_ + c] * Wgate[c];
    for (int c = 0; c < C_; ++c) s += sGE[tid * C_ + c] * Wgate[C_ + c];
    sGate[tid] = sigmoidf_(s);
  }
  __syncthreads();

  for (int idx = tid; idx < 64 * C_; idx += 256) {
    int m = idx >> 7, c = idx & 127;
    float pq = sPQ[idx] + sGate[m] * sGE[idx];
    pq_bf[((size_t)b * WH_ + p0 + m) * C_ + c] = (__bf16)pq;
  }
  for (int idx = tid; idx < 64 * C_; idx += 256) {
    int c = idx >> 6, m = idx & 63;
    geoT[((size_t)b * C_ + c) * WH_ + p0 + m] = sGE[m * C_ + c];
  }
}

// ---------------------------------------------------------------------------
// Kernel 2: pk = feat1^T @ Wk + bk (bf16 [b][n][c]);
//           pvT = (feat1^T @ Wv + bv)^T (bf16 [b][c][n]).
// ---------------------------------------------------------------------------
__global__ void proj_kv_kernel(const float* __restrict__ feat1,
                               const float* __restrict__ Wk, const float* __restrict__ bk,
                               const float* __restrict__ Wv, const float* __restrict__ bv,
                               __bf16* __restrict__ pk_bf, __bf16* __restrict__ pvT)
{
  extern __shared__ __align__(16) char smem[];
  __bf16* sWk  = (__bf16*)smem;         // [128][PITCH] transposed
  __bf16* sWv  = sWk + C_ * PITCH;
  __bf16* sA   = sWv + C_ * PITCH;      // kv_in tile [64][PITCH]
  __bf16* sKst = sA  + 64 * PITCH;      // [64][PITCH]
  __bf16* sVst = sKst + 64 * PITCH;     // [128][VPITCH]

  const int tid = threadIdx.x;
  const int b  = blockIdx.x / (N_ / 64);
  const int n0 = (blockIdx.x % (N_ / 64)) * 64;

  for (int idx = tid; idx < C_ * C_; idx += 256) {
    int k = idx >> 7, n = idx & 127;
    sWk[n * PITCH + k] = (__bf16)Wk[idx];
    sWv[n * PITCH + k] = (__bf16)Wv[idx];
  }
  for (int idx = tid; idx < 64 * C_; idx += 256) {
    int c = idx >> 6, m = idx & 63;
    sA[m * PITCH + c] = (__bf16)feat1[((size_t)b * C_ + c) * N_ + (n0 + m)];
  }

  const int w = tid >> 5, lane = tid & 31, nl = lane & 15;
  const bool is_k = (w < 4);
  const __bf16* Wmat = is_k ? sWk : sWv;
  const float*  bias = is_k ? bk  : bv;
  const int mrow = (w & 3) * 16;
  const int mb = mrow + ((lane & 16) ? 8 : 0);

  float bb[8];
#pragma unroll
  for (int ct = 0; ct < 8; ++ct) bb[ct] = bias[ct * 16 + nl];

  __syncthreads();

  const __bf16* arow = sA + (mrow + nl) * PITCH;
  bf16x16 afrag[4];
#pragma unroll
  for (int kc = 0; kc < 4; ++kc) afrag[kc] = ld_frag(arow, kc * 32, lane);

  v8f res[8];
  gemm_strip<false>(afrag, Wmat, nl, lane, res);

#pragma unroll
  for (int ct = 0; ct < 8; ++ct) {
    const int n = ct * 16 + nl;
    if (is_k) {
#pragma unroll
      for (int j = 0; j < 8; ++j)
        sKst[(mb + j) * PITCH + n] = (__bf16)(res[ct][j] + bb[ct]);
    } else {
#pragma unroll
      for (int j = 0; j < 8; ++j)
        sVst[n * VPITCH + mb + j] = (__bf16)(res[ct][j] + bb[ct]);
    }
  }
  __syncthreads();

  for (int i = tid; i < 64 * 16; i += 256) {
    int m = i >> 4, r = i & 15;
    *(uint4*)(pk_bf + ((size_t)b * N_ + n0 + m) * C_ + r * 8) =
        *(const uint4*)(sKst + m * PITCH + r * 8);
  }
  for (int i = tid; i < 128 * 8; i += 256) {
    int c = i >> 3, r = i & 7;
    *(uint4*)(pvT + ((size_t)b * C_ + c) * N_ + n0 + r * 8) =
        *(const uint4*)(sVst + c * VPITCH + r * 8);
  }
}

// ---------------------------------------------------------------------------
// Kernel 3: flash attention. 128 query rows per block. Wave tiling: 4 row
// groups x 2 column groups; each wave owns a 32-row x 64-col block so K/V
// B-fragments are loaded once and used by two WMMAs (halves LDS B traffic).
// Row softmax stats are combined across the column-group wave pair via LDS.
// KV double-buffered with GLOBAL_LOAD_ASYNC_TO_LDS.
// ---------------------------------------------------------------------------
__device__ __forceinline__ void stage_chunk(const __bf16* Kg, const __bf16* VTg,
                                            int n0, __bf16* sKb, __bf16* sVb, int tid)
{
  const char* gk = (const char*)(Kg + (size_t)n0 * C_);
  for (int i = tid; i < 2048; i += 256) {
    int m = i >> 4, r = i & 15;
    copy_b128(gk + i * 16, sKb + m * PITCH + r * 8);
  }
  for (int i = tid; i < 2048; i += 256) {
    int c = i >> 4, r = i & 15;
    copy_b128((const void*)(VTg + (size_t)c * N_ + n0 + r * 8),
              sVb + c * PITCH + r * 8);
  }
}

__global__ void flash_attn_kernel(const __bf16* __restrict__ pq_bf,
                                  const __bf16* __restrict__ pk_bf,
                                  const __bf16* __restrict__ pvT,
                                  __bf16* __restrict__ attn_bf)
{
  extern __shared__ __align__(16) char smem[];
  __bf16* sQ   = (__bf16*)smem;                // [128][PITCH]
  __bf16* sK0  = sQ  + 128 * PITCH;            // K double buffer
  __bf16* sK1  = sK0 + 128 * PITCH;
  __bf16* sV0  = sK1 + 128 * PITCH;            // V^T double buffer ([c][n])
  __bf16* sV1  = sV0 + 128 * PITCH;
  __bf16* sP   = sV1 + 128 * PITCH;            // [128][PITCH] shared P tile
  float*  sMax = (float*)(sP + 128 * PITCH);   // [2 colgrp][128 rows]
  float*  sSum = sMax + 256;                   // [2 colgrp][128 rows]

  const int tid = threadIdx.x;
  const int b  = blockIdx.x / (WH_ / 128);
  const int q0 = (blockIdx.x % (WH_ / 128)) * 128;
  const int w = tid >> 5, lane = tid & 31, nl = lane & 15;
  const int mb = (lane & 16) ? 8 : 0;
  const int rowgrp = w >> 1, colgrp = w & 1;
  const int mrow = rowgrp * 32;      // this wave's 32 query rows
  const int ctb  = colgrp * 4;       // this wave's 4 column tiles (of 8)

  const __bf16* Kg  = pk_bf + (size_t)b * N_ * C_;
  const __bf16* VTg = pvT   + (size_t)b * C_ * N_;

  // Stage Q (8 async ops/wave) + first KV chunk (16 ops/wave).
  {
    const char* gq = (const char*)(pq_bf + ((size_t)b * WH_ + q0) * C_);
    for (int i = tid; i < 2048; i += 256) {
      int m = i >> 4, r = i & 15;
      copy_b128(gq + i * 16, sQ + m * PITCH + r * 8);
    }
  }
  stage_chunk(Kg, VTg, 0, sK0, sV0, tid);

  float m_run[2][8], l_run[2][8];
  v8f o[2][4];
  const v8f vzero = {};
#pragma unroll
  for (int mt = 0; mt < 2; ++mt)
#pragma unroll
    for (int j = 0; j < 8; ++j) { m_run[mt][j] = -3.0e38f; l_run[mt][j] = 0.0f; }
#pragma unroll
  for (int mt = 0; mt < 2; ++mt)
#pragma unroll
    for (int ct = 0; ct < 4; ++ct) o[mt][ct] = vzero;

  bf16x16 qf[2][4];
  bool qloaded = false;

  const int NC = N_ / 128;
  for (int nc = 0; nc < NC; ++nc) {
    const int cur = nc & 1;
    const int nnext = ((nc + 1) < NC) ? (nc + 1) * 128 : 0;
    stage_chunk(Kg, VTg, nnext, cur ? sK0 : sK1, cur ? sV0 : sV1, tid);
    async_wait_16();
    __syncthreads();

    __bf16* sK = cur ? sK1 : sK0;
    __bf16* sV = cur ? sV1 : sV0;

    if (!qloaded) {   // Q fragments are loop-invariant: load once.
#pragma unroll
      for (int mt = 0; mt < 2; ++mt)
#pragma unroll
        for (int kc = 0; kc < 4; ++kc)
          qf[mt][kc] = ld_frag(sQ + (mrow + mt * 16 + nl) * PITCH, kc * 32, lane);
      qloaded = true;
    }

    // S = Q @ K^T: 32 rows x 64 cols; each B fragment feeds two WMMAs.
    v8f s[2][4];
#pragma unroll
    for (int ct = 0; ct < 4; ++ct) {
      const __bf16* brow = sK + ((ctb + ct) * 16 + nl) * PITCH;
      v8f a0 = vzero, a1 = vzero;
#pragma unroll
      for (int kc = 0; kc < 4; ++kc) {
        bf16x16 bfr = ld_frag(brow, kc * 32, lane);
        a0 = wmma_bf16(qf[0][kc], bfr, a0);
        a1 = wmma_bf16(qf[1][kc], bfr, a1);
      }
      s[0][ct] = a0;
      s[1][ct] = a1;
    }

    // Partial row max over this wave's 64 cols, then combine with partner.
    float pmax[2][8], alpha[2][8], psum[2][8];
#pragma unroll
    for (int mt = 0; mt < 2; ++mt)
#pragma unroll
      for (int j = 0; j < 8; ++j) {
        float v = s[mt][0][j];
#pragma unroll
        for (int ct = 1; ct < 4; ++ct) v = fmaxf(v, s[mt][ct][j]);
#pragma unroll
        for (int off = 1; off < 16; off <<= 1) v = fmaxf(v, __shfl_xor(v, off, 32));
        pmax[mt][j] = v;
        if (nl == 0) sMax[colgrp * 128 + mrow + mt * 16 + mb + j] = v;
      }
    __syncthreads();
#pragma unroll
    for (int mt = 0; mt < 2; ++mt)
#pragma unroll
      for (int j = 0; j < 8; ++j) {
        float om = sMax[(colgrp ^ 1) * 128 + mrow + mt * 16 + mb + j];
        float nm = fmaxf(m_run[mt][j], fmaxf(pmax[mt][j], om));
        alpha[mt][j] = __expf(m_run[mt][j] - nm);
        m_run[mt][j] = nm;
      }

    // exp + partial row sums; combine sums with partner.
#pragma unroll
    for (int mt = 0; mt < 2; ++mt)
#pragma unroll
      for (int j = 0; j < 8; ++j) {
        float acc = 0.0f;
#pragma unroll
        for (int ct = 0; ct < 4; ++ct) {
          float p = __expf(s[mt][ct][j] - m_run[mt][j]);
          s[mt][ct][j] = p;
          acc += p;
        }
#pragma unroll
        for (int off = 1; off < 16; off <<= 1) acc += __shfl_xor(acc, off, 32);
        psum[mt][j] = acc;
        if (nl == 0) sSum[colgrp * 128 + mrow + mt * 16 + mb + j] = acc;
      }
    __syncthreads();
#pragma unroll
    for (int mt = 0; mt < 2; ++mt)
#pragma unroll
      for (int j = 0; j < 8; ++j) {
        float tot = psum[mt][j] + sSum[(colgrp ^ 1) * 128 + mrow + mt * 16 + mb + j];
        l_run[mt][j] = l_run[mt][j] * alpha[mt][j] + tot;
#pragma unroll
        for (int ct = 0; ct < 4; ++ct) o[mt][ct][j] *= alpha[mt][j];
      }

    // P (accumulator layout) -> shared bf16 tile [row][kv-col].
#pragma unroll
    for (int mt = 0; mt < 2; ++mt)
#pragma unroll
      for (int ct = 0; ct < 4; ++ct)
#pragma unroll
        for (int j = 0; j < 8; ++j)
          sP[(mrow + mt * 16 + mb + j) * PITCH + (ctb + ct) * 16 + nl] =
              (__bf16)s[mt][ct][j];
    __syncthreads();   // partner's 64 columns of P are needed below

    // O += P @ V: P A-fragments (full 128-col K dim) loaded once per chunk;
    // V B-fragments loaded once and used by both M tiles.
    bf16x16 pf[2][4];
#pragma unroll
    for (int mt = 0; mt < 2; ++mt)
#pragma unroll
      for (int kc = 0; kc < 4; ++kc)
        pf[mt][kc] = ld_frag(sP + (mrow + mt * 16 + nl) * PITCH, kc * 32, lane);
#pragma unroll
    for (int ct = 0; ct < 4; ++ct) {
      const __bf16* brow = sV + ((ctb + ct) * 16 + nl) * PITCH;
      v8f a0 = o[0][ct], a1 = o[1][ct];
#pragma unroll
      for (int kc = 0; kc < 4; ++kc) {
        bf16x16 bfr = ld_frag(brow, kc * 32, lane);
        a0 = wmma_bf16(pf[0][kc], bfr, a0);
        a1 = wmma_bf16(pf[1][kc], bfr, a1);
      }
      o[0][ct] = a0;
      o[1][ct] = a1;
    }
    __syncthreads();
  }

  // Drain the wrap-around prefetch before reusing sK0 as output staging.
  async_wait_all();
  __syncthreads();

  // attention = softmax(energy)/sqrt(C): scale rows by 1/(l*sqrt(128)),
  // stage bf16 result in LDS, then b128 coalesced stores.
  const float rsC = 0.08838834764831845f; // 1/sqrt(128)
  __bf16* sOutA = sK0;                    // reuse: [128][PITCH]
#pragma unroll
  for (int mt = 0; mt < 2; ++mt) {
    float inv[8];
#pragma unroll
    for (int j = 0; j < 8; ++j) inv[j] = rsC / l_run[mt][j];
#pragma unroll
    for (int ct = 0; ct < 4; ++ct)
#pragma unroll
      for (int j = 0; j < 8; ++j)
        sOutA[(mrow + mt * 16 + mb + j) * PITCH + (ctb + ct) * 16 + nl] =
            (__bf16)(o[mt][ct][j] * inv[j]);
  }
  __syncthreads();
  for (int i = tid; i < 2048; i += 256) {
    int m = i >> 4, r = i & 15;
    *(uint4*)(attn_bf + ((size_t)b * WH_ + q0 + m) * C_ + r * 8) =
        *(const uint4*)(sOutA + m * PITCH + r * 8);
  }
}

// ---------------------------------------------------------------------------
// Kernel 4: out = (attn @ Wo + bo) * (1 + sigmoid(geo)) + feat0, stored [b][c][p].
// ---------------------------------------------------------------------------
__global__ void out_proj_kernel(const __bf16* __restrict__ attn_bf,
                                const float* __restrict__ Wo, const float* __restrict__ bo,
                                const float* __restrict__ geoT,
                                const float* __restrict__ feat0,
                                float* __restrict__ out)
{
  extern __shared__ __align__(16) char smem[];
  __bf16* sWo  = (__bf16*)smem;              // [128][PITCH] transposed
  __bf16* sA   = sWo + C_ * PITCH;           // [128][PITCH]
  float*  sOut = (float*)(sA + C_ * PITCH);  // [128 c][OPITCH] fp32

  const int tid = threadIdx.x;
  const int b  = blockIdx.x / (WH_ / 128);
  const int q0 = (blockIdx.x % (WH_ / 128)) * 128;

  for (int idx = tid; idx < C_ * C_; idx += 256) {
    int k = idx >> 7, n = idx & 127;
    sWo[n * PITCH + k] = (__bf16)Wo[idx];
  }
  {
    const char* g = (const char*)(attn_bf + ((size_t)b * WH_ + q0) * C_);
    for (int i = tid; i < 2048; i += 256) {
      int m = i >> 4, r = i & 15;
      copy_b128(g + i * 16, sA + m * PITCH + r * 8);
    }
  }

  const int w = tid >> 5, lane = tid & 31, nl = lane & 15;
  const int mb = (lane & 16) ? 8 : 0;

  float bb[8];
#pragma unroll
  for (int ct = 0; ct < 8; ++ct) bb[ct] = bo[ct * 16 + nl];

  async_wait_all();
  __syncthreads();

  const __bf16* arow = sA + (w * 16 + nl) * PITCH;
  bf16x16 afrag[4];
#pragma unroll
  for (int kc = 0; kc < 4; ++kc) afrag[kc] = ld_frag(arow, kc * 32, lane);

  v8f res[8];
  gemm_strip<false>(afrag, sWo, nl, lane, res);
#pragma unroll
  for (int ct = 0; ct < 8; ++ct)
#pragma unroll
    for (int j = 0; j < 8; ++j)
      sOut[(ct * 16 + nl) * OPITCH + w * 16 + mb + j] = res[ct][j] + bb[ct];
  __syncthreads();

  for (int i = tid; i < 128 * 32; i += 256) {
    int c = i >> 5, r = i & 31;
    const size_t base = ((size_t)b * C_ + c) * WH_ + q0 + r * 4;
    float4 v = *(const float4*)(sOut + c * OPITCH + r * 4);
    float4 g = *(const float4*)(geoT + base);
    float4 f = *(const float4*)(feat0 + base);
    v.x = v.x * (1.0f + sigmoidf_(g.x)) + f.x;
    v.y = v.y * (1.0f + sigmoidf_(g.y)) + f.y;
    v.z = v.z * (1.0f + sigmoidf_(g.z)) + f.z;
    v.w = v.w * (1.0f + sigmoidf_(g.w)) + f.w;
    *(float4*)(out + base) = v;
  }
}

// ---------------------------------------------------------------------------
extern "C" void kernel_launch(void* const* d_in, const int* in_sizes, int n_in,
                              void* d_out, int out_size, void* d_ws, size_t ws_size,
                              hipStream_t stream)
{
  (void)in_sizes; (void)n_in; (void)out_size; (void)ws_size;
  const float* feat0     = (const float*)d_in[0];
  const float* feat1     = (const float*)d_in[1];
  const float* geo_embed = (const float*)d_in[2];
  const float* Wq = (const float*)d_in[3];   const float* bq = (const float*)d_in[4];
  const float* Wk = (const float*)d_in[5];   const float* bk = (const float*)d_in[6];
  const float* Wv = (const float*)d_in[7];   const float* bv = (const float*)d_in[8];
  const float* Wo = (const float*)d_in[9];   const float* bo = (const float*)d_in[10];
  const float* Wg = (const float*)d_in[11];  const float* bg = (const float*)d_in[12];
  const float* Wgate = (const float*)d_in[13]; const float* bgate = (const float*)d_in[14];
  float* out = (float*)d_out;

  char* ws = (char*)d_ws;
  size_t off = 0;
  auto alloc = [&](size_t bytes) -> char* {
    char* p = ws + off;
    off = (off + bytes + 255) & ~(size_t)255;
    return p;
  };
  __bf16* pq_bf   = (__bf16*)alloc((size_t)B_ * WH_ * C_ * 2);
  __bf16* pk_bf   = (__bf16*)alloc((size_t)B_ * N_  * C_ * 2);
  __bf16* pvT     = (__bf16*)alloc((size_t)B_ * C_  * N_ * 2);
  float*  geoT    = (float* )alloc((size_t)B_ * C_  * WH_ * 4);
  __bf16* attn_bf = (__bf16*)alloc((size_t)B_ * WH_ * C_ * 2);

  const size_t smem_qgeo = (size_t)(2 * C_ * PITCH * 2) + (2 * 64 * PITCH * 2)
                         + (2 * 64 * C_ * 4) + 64 * 4;
  proj_qgeo_kernel<<<B_ * (WH_ / 64), 256, smem_qgeo, stream>>>(
      feat0, geo_embed, Wq, bq, Wg, bg, Wgate, bgate, pq_bf, geoT);

  const size_t smem_kv = (size_t)(2 * C_ * PITCH * 2) + (2 * 64 * PITCH * 2)
                       + (size_t)C_ * VPITCH * 2;
  proj_kv_kernel<<<B_ * (N_ / 64), 256, smem_kv, stream>>>(
      feat1, Wk, bk, Wv, bv, pk_bf, pvT);

  const size_t smem_attn = (size_t)6 * 128 * PITCH * 2 + 512 * 4;
  flash_attn_kernel<<<B_ * (WH_ / 128), 256, smem_attn, stream>>>(
      pq_bf, pk_bf, pvT, attn_bf);

  const size_t smem_out = (size_t)(2 * C_ * PITCH * 2) + (size_t)C_ * OPITCH * 4;
  out_proj_kernel<<<B_ * (WH_ / 128), 256, smem_out, stream>>>(
      attn_bf, Wo, bo, geoT, feat0, out);
}
